// basket_GRU_76295799046562
// MI455X (gfx1250) — matrix operand order repaired and checked by reading, not compiled, not run on hardware
//
#include <hip/hip_runtime.h>
#include <math.h>

// ---------------------------------------------------------------------------
// basket-GRU for MI455X (gfx1250, wave32, WMMA bf16 -> f32 accumulate)
//
// All GEMM operands are pre-swizzled into the exact WMMA fragment order of
// CDNA5 ISA 7.12.2, so every fragment load is one contiguous 32B load/lane.
//   A tiles (16x32 bf16): tile (tm,tk) at ((tm*(K/32)+tk)*512), lane*16+e
//   B tiles (32x16 bf16): tile (tk,tn) at ((tk*(N/16)+tn)*512), lane*16+e
//
// proj_gemm_kernel stages the block-shared A row-tile (32 KB) into LDS via the
// Tensor Data Mover.  All GEMM loops use split-K dual accumulators so
// consecutive v_wmma instructions are independent (no 5-NOP RAW bubbles).
// ---------------------------------------------------------------------------

typedef __attribute__((ext_vector_type(16))) __bf16        v16bf;
typedef __attribute__((ext_vector_type(8)))  float         v8f;
typedef __attribute__((ext_vector_type(4)))  unsigned int  v4u;
typedef __attribute__((ext_vector_type(8)))  int           v8i_;
typedef __attribute__((ext_vector_type(4)))  int           v4i;

#define BB 128
#define TT 200
#define DD 1024
#define HH 512

// ---- helpers ---------------------------------------------------------------

__device__ __forceinline__ unsigned short f2bf(float f) {
  unsigned int u = __float_as_uint(f);                 // RNE truncate to bf16
  unsigned int r = u + 0x7FFFu + ((u >> 16) & 1u);
  return (unsigned short)(r >> 16);
}
__device__ __forceinline__ float sigmoidf_(float x) {
  return 1.0f / (1.0f + __expf(-x));
}

// index of element (row,k) inside an A-fragment-ordered [M x K] bf16 buffer
__device__ __forceinline__ size_t a_swz_idx(int row, int k, int K) {
  int tm = row >> 4, tk = k >> 5;
  int rl = row & 15, kl = k & 31;
  int half = (kl >> 3) & 1;                    // k {0-7,16-23}->lo, {8-15,24-31}->hi
  int lane = rl + (half << 4);
  int e    = (kl & 7) + ((kl >= 16) ? 8 : 0);
  return ((size_t)(tm * (K >> 5) + tk)) * 512 + (size_t)lane * 16 + e;
}

// 16x16 f32 tile of A[MxK] @ B[KxN], both in swizzled fragment order.
// Split-K dual accumulators: alternating WMMAs are independent, hiding the
// WMMA->WMMA accumulator RAW hazard (ISA 7.12.1).
__device__ __forceinline__ v8f wmma_tile_swz(const __bf16* __restrict__ A,
                                             const __bf16* __restrict__ Bm,
                                             int tm, int tn, int K, int N, int lane) {
  const int tilesK = K >> 5;                   // even for K in {512, 1024}
  const int tilesN = N >> 4;
  const __bf16* ap = A  + ((size_t)tm * tilesK) * 512 + (size_t)lane * 16;
  const __bf16* bp = Bm + ((size_t)tn) * 512 + (size_t)lane * 16;
  v8f c0 = {}, c1 = {};
  for (int tk = 0; tk < tilesK; tk += 2) {
    if (tk + 2 < tilesK) {
      __builtin_prefetch(bp + (size_t)(tk + 2) * tilesN * 512, 0, 1);
    }
    v16bf a0 = *(const v16bf*)(ap + (size_t)tk * 512);
    v16bf b0 = *(const v16bf*)(bp + (size_t)tk * tilesN * 512);
    v16bf a1 = *(const v16bf*)(ap + (size_t)(tk + 1) * 512);
    v16bf b1 = *(const v16bf*)(bp + (size_t)(tk + 1) * tilesN * 512);
    c0 = __builtin_amdgcn_wmma_f32_16x16x32_bf16(false, a0, false, b0, (short)0, c0,
                                                 false, false);
    c1 = __builtin_amdgcn_wmma_f32_16x16x32_bf16(false, a1, false, b1, (short)0, c1,
                                                 false, false);
  }
  return c0 + c1;
}

// atomic grid barrier for the persistent recurrent kernel
__device__ __forceinline__ void grid_barrier(unsigned int* cnt, unsigned int* gen,
                                             unsigned int nblk) {
  __syncthreads();
  if (threadIdx.x == 0) {
    __threadfence();
    unsigned int g = __hip_atomic_load(gen, __ATOMIC_RELAXED, __HIP_MEMORY_SCOPE_AGENT);
    unsigned int a = __hip_atomic_fetch_add(cnt, 1u, __ATOMIC_ACQ_REL,
                                            __HIP_MEMORY_SCOPE_AGENT);
    if (a == nblk - 1u) {
      __hip_atomic_store(cnt, 0u, __ATOMIC_RELAXED, __HIP_MEMORY_SCOPE_AGENT);
      __hip_atomic_store(gen, g + 1u, __ATOMIC_RELEASE, __HIP_MEMORY_SCOPE_AGENT);
    } else {
      while (__hip_atomic_load(gen, __ATOMIC_ACQUIRE, __HIP_MEMORY_SCOPE_AGENT) == g) {
        __builtin_amdgcn_s_sleep(2);
      }
    }
  }
  __syncthreads();
}

// ---- swizzling conversion kernels ------------------------------------------

// f32 [M x K] row-major -> bf16 A-fragment order (coalesced writes)
__global__ void swzA_kernel(const float* __restrict__ src, unsigned short* __restrict__ dst,
                            int M, int K) {
  int i = blockIdx.x * blockDim.x + threadIdx.x;
  if (i >= M * K) return;
  int tile = i >> 9, pos = i & 511;
  int lane = pos >> 4, e = pos & 15;
  int tilesK = K >> 5;
  int tm = tile / tilesK, tk = tile % tilesK;
  int row = tm * 16 + (lane & 15);
  int kl  = ((e < 8) ? e : e + 8) + ((lane >> 4) << 3);
  dst[i] = f2bf(src[(size_t)row * K + (tk * 32 + kl)]);
}

// f32 [K x N] row-major -> bf16 B-fragment order (coalesced writes)
__global__ void swzB_kernel(const float* __restrict__ src, unsigned short* __restrict__ dst,
                            int K, int N) {
  int i = blockIdx.x * blockDim.x + threadIdx.x;
  if (i >= K * N) return;
  int tile = i >> 9, pos = i & 511;
  int lane = pos >> 4, e = pos & 15;
  int tilesN = N >> 4;
  int tk = tile / tilesN, tn = tile % tilesN;
  int col = tn * 16 + (lane & 15);
  int k   = tk * 32 + ((e < 8) ? e : e + 8) + ((lane >> 4) << 3);
  dst[i] = f2bf(src[(size_t)k * N + col]);
}

// ---- projection GEMM: xf/xs/xi = x @ W + bias ------------------------------
// 8 waves/block share one A row-tile (16 rows x 1024 K = 32 KB bf16), staged
// into LDS with the Tensor Data Mover; each wave owns one 16x16 output tile.
__global__ void proj_gemm_kernel(const unsigned short* __restrict__ xbf,  // A swizzled
                                 const unsigned short* __restrict__ Wfb,
                                 const unsigned short* __restrict__ Wsb,
                                 const unsigned short* __restrict__ Wib,
                                 const float* __restrict__ ef,
                                 const float* __restrict__ es,
                                 const float* __restrict__ ei,
                                 float* __restrict__ xf,
                                 float* __restrict__ xs,
                                 float* __restrict__ xi) {
  __shared__ __bf16 shA[16384];  // 32 KB: 32 k-tiles x 512 elements

  const int TILES_N = HH / 16;                         // 32
  const int tilesPerGate = (BB * TT / 16) * TILES_N;   // 51200
  const int wave0 = blockIdx.x * 8;
  const int gate  = wave0 / tilesPerGate;              // uniform per block
  const int idx0  = wave0 % tilesPerGate;
  const int tm    = idx0 / TILES_N;                    // uniform per block
  const int tn0   = idx0 % TILES_N;

  // Visible stores into shA so the compiler cannot treat it as never-written
  // (the TDM write below is opaque to alias analysis).
  shA[threadIdx.x] = (__bf16)0.0f;

  // --- TDM: copy this block's A row-tile (contiguous 16384 bf16) to LDS ---
  if (threadIdx.x < 32) {   // wave 0 issues once (EXEC is ignored by TDM)
    unsigned long long ga =
        (unsigned long long)(uintptr_t)(xbf + (size_t)tm * 16384);
    v4u g0;
    g0[0] = 1u;                                        // count=1, user mode
    g0[1] = 0u;                                        // lds_addr: shA at LDS base
    g0[2] = (unsigned int)(ga & 0xFFFFFFFFu);          // global_addr[31:0]
    g0[3] = (unsigned int)((ga >> 32) & 0x1FFFFFFu) | (2u << 30);  // addr hi | type=2
    v8i_ g1;
    g1[0] = (int)(1u << 16);        // workgroup_mask=0, data_size=1 (2 bytes)
    g1[1] = (int)(16384u << 16);    // tensor_dim0[15:0] in [31:16]  (=16384)
    g1[2] = (int)(1u << 16);        // tensor_dim0 hi=0; tensor_dim1=1
    g1[3] = (int)(16384u << 16);    // tensor_dim1 hi=0; tile_dim0=16384
    g1[4] = 0;                      // tile_dim1=0 (unused), tile_dim2=0
    g1[5] = 16384;                  // tensor_dim0_stride lo
    g1[6] = 0;
    g1[7] = 0;
    v4i  z4 = {0, 0, 0, 0};
    v8i_ z8 = {0, 0, 0, 0, 0, 0, 0, 0};
    __builtin_amdgcn_tensor_load_to_lds(g0, g1, z4, z4, z8, 0);
    __builtin_amdgcn_s_wait_tensorcnt(0);
  }
  __syncthreads();
  // Escape shA's address + memory clobber: the TDM rewrote LDS behind the
  // compiler's back, so force real ds_loads below.
  asm volatile("" : : "v"((unsigned int)(size_t)(void*)shA) : "memory");

  const int lane = threadIdx.x & 31;
  const int wv   = threadIdx.x >> 5;
  const int tn   = tn0 + wv;

  const unsigned short* Wt = (gate == 0) ? Wfb : (gate == 1) ? Wsb : Wib;
  const float* bias        = (gate == 0) ? ef  : (gate == 1) ? es  : ei;
  float* op                = (gate == 0) ? xf  : (gate == 1) ? xs  : xi;

  const __bf16* ap = shA + (size_t)lane * 16;
  const __bf16* bp = (const __bf16*)Wt + (size_t)tn * 512 + (size_t)lane * 16;

  v8f c0 = {}, c1 = {};
  for (int tk = 0; tk < DD / 32; tk += 2) {            // 32 k-tiles, dual accum
    if (tk + 2 < DD / 32) {
      __builtin_prefetch(bp + (size_t)(tk + 2) * TILES_N * 512, 0, 1);
    }
    v16bf a0 = *(const v16bf*)(ap + (size_t)tk * 512);               // ds_load
    v16bf b0 = *(const v16bf*)(bp + (size_t)tk * TILES_N * 512);     // global_load
    v16bf a1 = *(const v16bf*)(ap + (size_t)(tk + 1) * 512);
    v16bf b1 = *(const v16bf*)(bp + (size_t)(tk + 1) * TILES_N * 512);
    c0 = __builtin_amdgcn_wmma_f32_16x16x32_bf16(false, a0, false, b0, (short)0, c0,
                                                 false, false);
    c1 = __builtin_amdgcn_wmma_f32_16x16x32_bf16(false, a1, false, b1, (short)0, c1,
                                                 false, false);
  }
  v8f c = c0 + c1;

  const int col  = tn * 16 + (lane & 15);
  const int row0 = tm * 16 + ((lane >> 4) << 3);
  const float bv = bias[col];
#pragma unroll
  for (int r = 0; r < 8; ++r)
    op[(size_t)(row0 + r) * HH + col] = c[r] + bv;
}

// ---- persistent recurrent scan ---------------------------------------------
// 64 blocks x 256 threads = 512 wave32 waves; 3 phases / step, grid barrier
// between phases.  hbf / hmod live in A-fragment-swizzled bf16 buffers.
__global__ void recurrent_kernel(const float* __restrict__ xf,
                                 const float* __restrict__ xs,
                                 const float* __restrict__ xi,
                                 const unsigned short* __restrict__ Wfa,
                                 const unsigned short* __restrict__ Wsa,
                                 const unsigned short* __restrict__ Wia,
                                 const unsigned short* __restrict__ Wba,
                                 const float* __restrict__ e_o,
                                 float* __restrict__ h,
                                 unsigned short* __restrict__ hbf,   // A-swizzled
                                 unsigned short* __restrict__ hmod,  // A-swizzled
                                 float* __restrict__ Fg,
                                 float* __restrict__ out,
                                 unsigned int* cnt, unsigned int* gen) {
  const int lane = threadIdx.x & 31;
  const int gw   = (int)((blockIdx.x * (unsigned)blockDim.x + threadIdx.x) >> 5); // 0..511
  const unsigned int nblk = gridDim.x;

  const __bf16* hbf_b  = (const __bf16*)hbf;
  const __bf16* hmod_b = (const __bf16*)hmod;

  for (int t = 0; t < TT; ++t) {
    // ---- Phase A: forget (waves 0..255) & modulator (waves 256..511) ----
    {
      const int idx  = gw & 255;
      const int tm   = idx >> 5;
      const int tn   = idx & 31;
      const int col  = tn * 16 + (lane & 15);
      const int row0 = tm * 16 + ((lane >> 4) << 3);
      if (gw < 256) {
        v8f c = wmma_tile_swz(hbf_b, (const __bf16*)Wfa, tm, tn, HH, HH, lane);
#pragma unroll
        for (int r = 0; r < 8; ++r) {
          const int row = row0 + r;
          float pre = c[r] + xf[((size_t)row * TT + t) * HH + col];
          Fg[(size_t)row * HH + col] = sigmoidf_(pre);
        }
      } else {
        v8f c = wmma_tile_swz(hbf_b, (const __bf16*)Wsa, tm, tn, HH, HH, lane);
#pragma unroll
        for (int r = 0; r < 8; ++r) {
          const int row = row0 + r;
          float s  = sigmoidf_(c[r] + xs[((size_t)row * TT + t) * HH + col]);
          hmod[a_swz_idx(row, col, HH)] = f2bf(h[(size_t)row * HH + col] * s);
        }
      }
    }
    grid_barrier(cnt, gen, nblk);

    // ---- Phase B: candidate + state update (waves 0..255) ----
    if (gw < 256) {
      const int tm   = gw >> 5;
      const int tn   = gw & 31;
      const int col  = tn * 16 + (lane & 15);
      const int row0 = tm * 16 + ((lane >> 4) << 3);
      v8f c = wmma_tile_swz(hmod_b, (const __bf16*)Wia, tm, tn, HH, HH, lane);
#pragma unroll
      for (int r = 0; r < 8; ++r) {
        const int row = row0 + r;
        float cand = tanhf(c[r] + xi[((size_t)row * TT + t) * HH + col]);
        float f  = Fg[(size_t)row * HH + col];
        float hv = h[(size_t)row * HH + col];
        float hn = hv * (1.0f - f) + f * cand;
        h[(size_t)row * HH + col]  = hn;
        hbf[a_swz_idx(row, col, HH)] = f2bf(hn);
      }
    }
    grid_barrier(cnt, gen, nblk);

    // ---- Phase C: output projection (all 512 waves, 8x64 tiles) ----
    {
      const int tm   = gw >> 6;
      const int tn   = gw & 63;
      const int col  = tn * 16 + (lane & 15);
      const int row0 = tm * 16 + ((lane >> 4) << 3);
      v8f c = wmma_tile_swz(hbf_b, (const __bf16*)Wba, tm, tn, HH, DD, lane);
      const float bo = e_o[col];
#pragma unroll
      for (int r = 0; r < 8; ++r) {
        const int row = row0 + r;
        out[((size_t)row * TT + t) * DD + col] = sigmoidf_(c[r] + bo);
      }
    }
    grid_barrier(cnt, gen, nblk);
  }
}

// ---- host launch ------------------------------------------------------------

extern "C" void kernel_launch(void* const* d_in, const int* in_sizes, int n_in,
                              void* d_out, int out_size, void* d_ws, size_t ws_size,
                              hipStream_t stream) {
  const float* x    = (const float*)d_in[0];
  const float* W_fb = (const float*)d_in[1];
  const float* W_fa = (const float*)d_in[2];
  const float* e_f  = (const float*)d_in[3];
  const float* W_ib = (const float*)d_in[4];
  const float* W_ia = (const float*)d_in[5];
  const float* e_i  = (const float*)d_in[6];
  const float* W_sb = (const float*)d_in[7];
  const float* W_sa = (const float*)d_in[8];
  const float* e_s  = (const float*)d_in[9];
  const float* W_ba = (const float*)d_in[10];
  const float* e_o  = (const float*)d_in[11];
  float* out = (float*)d_out;

  char* w = (char*)d_ws;
  auto carve = [&](size_t bytes) -> char* {
    char* p = w;
    w += (bytes + 255) & ~(size_t)255;
    return p;
  };
  const size_t MT = (size_t)BB * TT;  // 25600 rows
  float* xf = (float*)carve(MT * HH * sizeof(float));
  float* xs = (float*)carve(MT * HH * sizeof(float));
  float* xi = (float*)carve(MT * HH * sizeof(float));
  unsigned short* xbf   = (unsigned short*)carve(MT * DD * 2);      // A-swizzled x
  unsigned short* Wfb_b = (unsigned short*)carve((size_t)DD * HH * 2);
  unsigned short* Wsb_b = (unsigned short*)carve((size_t)DD * HH * 2);
  unsigned short* Wib_b = (unsigned short*)carve((size_t)DD * HH * 2);
  unsigned short* Wfa_b = (unsigned short*)carve((size_t)HH * HH * 2);
  unsigned short* Wsa_b = (unsigned short*)carve((size_t)HH * HH * 2);
  unsigned short* Wia_b = (unsigned short*)carve((size_t)HH * HH * 2);
  unsigned short* Wba_b = (unsigned short*)carve((size_t)HH * DD * 2);
  float*          h     = (float*)carve((size_t)BB * HH * 4);
  unsigned short* hbf   = (unsigned short*)carve((size_t)BB * HH * 2);
  unsigned short* hmod  = (unsigned short*)carve((size_t)BB * HH * 2);
  float*          Fg    = (float*)carve((size_t)BB * HH * 4);
  unsigned int*   bar   = (unsigned int*)carve(256);  // [0]=count, [32]=generation

  // per-launch init (deterministic across graph replays)
  (void)hipMemsetAsync(h,   0, (size_t)BB * HH * 4, stream);
  (void)hipMemsetAsync(hbf, 0, (size_t)BB * HH * 2, stream);
  (void)hipMemsetAsync(bar, 0, 256, stream);

  // swizzle x (A operand) and weights (B operands) into fragment order
  {
    int n = (int)(MT * DD);
    swzA_kernel<<<(n + 255) / 256, 256, 0, stream>>>(x, xbf, (int)MT, DD);
  }
  auto swzB = [&](const float* s, unsigned short* d, int K, int N) {
    int n = K * N;
    swzB_kernel<<<(n + 255) / 256, 256, 0, stream>>>(s, d, K, N);
  };
  swzB(W_fb, Wfb_b, DD, HH);
  swzB(W_sb, Wsb_b, DD, HH);
  swzB(W_ib, Wib_b, DD, HH);
  swzB(W_fa, Wfa_b, HH, HH);
  swzB(W_sa, Wsa_b, HH, HH);
  swzB(W_ia, Wia_b, HH, HH);
  swzB(W_ba, Wba_b, HH, DD);

  // Phase 1: input projections (3 gates x 1600x32 tiles, 8 waves/block,
  //          block-shared A tile staged via TDM)
  {
    const int totalWaves = 3 * (BB * TT / 16) * (HH / 16);  // 153600
    const int blocks = totalWaves / 8;                      // 19200
    proj_gemm_kernel<<<blocks, 256, 0, stream>>>(xbf, Wfb_b, Wsb_b, Wib_b,
                                                 e_f, e_s, e_i, xf, xs, xi);
  }

  // Phase 2: persistent recurrent scan (64 blocks co-resident, grid barrier)
  recurrent_kernel<<<64, 256, 0, stream>>>(xf, xs, xi, Wfa_b, Wsa_b, Wia_b, Wba_b,
                                           e_o, h, hbf, hmod, Fg, out,
                                           bar, bar + 32);
}
